// SynthesisBlock_20882130993777
// MI455X (gfx1250) — compile-verified, matrix operand order checked by reading, not run on hardware
//
#include <hip/hip_runtime.h>
#include <cmath>

typedef __attribute__((ext_vector_type(8)))  _Float16 v8h;
typedef __attribute__((ext_vector_type(16))) _Float16 v16h;
typedef __attribute__((ext_vector_type(8)))  float    v8f;

#define NB    8      // batch
#define SDIM  512    // style dim
#define COUT  256
#define HOUT  128    // output spatial (after 2x up)
#define HIN   64

// ---------------------------------------------------------------------------
// Weight prep: w[o][c][3][3] f32  ->  wh[tap][o][c] f16 (scale folded) + wsq[o][c]
// ---------------------------------------------------------------------------
template<int CIN>
__global__ void prep_weights_kernel(const float* __restrict__ w,
                                    _Float16* __restrict__ wh,
                                    float* __restrict__ wsq, float scale) {
    int idx = blockIdx.x * 256 + threadIdx.x;          // o*CIN + c
    if (idx >= COUT * CIN) return;
    int o = idx / CIN, c = idx % CIN;
    const float* src = w + (size_t)(o * CIN + c) * 9;
    float sq = 0.f;
#pragma unroll
    for (int t = 0; t < 9; ++t) {
        float v = src[t];
        sq += v * v;
        wh[((size_t)t * COUT + o) * CIN + c] = (_Float16)(v * scale);
    }
    wsq[idx] = sq;
}

// ---------------------------------------------------------------------------
// EqualLinear styles: s = style @ (W/sqrt(S))^T + b
// ---------------------------------------------------------------------------
__global__ void styles_kernel(const float* __restrict__ w1v, const float* __restrict__ w2v,
                              const float* __restrict__ m1w, const float* __restrict__ m1b,
                              const float* __restrict__ m2w, const float* __restrict__ m2b,
                              const float* __restrict__ m3w, const float* __restrict__ m3b,
                              float* __restrict__ s1, float* __restrict__ s2,
                              float* __restrict__ s3) {
    __shared__ float a1[SDIM], a2[SDIM];
    int b = blockIdx.x, tid = threadIdx.x;
    for (int i = tid; i < SDIM; i += 256) { a1[i] = w1v[b * SDIM + i]; a2[i] = w2v[b * SDIM + i]; }
    __syncthreads();
    const float inv = 0.04419417382415922f;            // 1/sqrt(512)
    for (int i = tid; i < SDIM; i += 256) {            // s1: 512 channels
        float acc = 0.f;
        for (int j = 0; j < SDIM; ++j) acc += a1[j] * m1w[(size_t)i * SDIM + j];
        s1[b * SDIM + i] = acc * inv + m1b[i];
    }
    {                                                  // s2/s3: 256 channels
        int i = tid;
        float acc2 = 0.f, acc3 = 0.f;
        for (int j = 0; j < SDIM; ++j) {
            float av = a2[j];
            acc2 += av * m2w[(size_t)i * SDIM + j];
            acc3 += av * m3w[(size_t)i * SDIM + j];
        }
        s2[b * COUT + i] = acc2 * inv + m2b[i];
        s3[b * COUT + i] = acc3 * inv + m3b[i];
    }
}

// ---------------------------------------------------------------------------
// Demodulation: d[b][o] = rsqrt(scale^2 * sum_c wsq[o][c]*s[b][c]^2 + 1e-8)
// ---------------------------------------------------------------------------
__global__ void demod_kernel(const float* __restrict__ wsq1, const float* __restrict__ wsq2,
                             const float* __restrict__ s1,  const float* __restrict__ s2,
                             float* __restrict__ d1, float* __restrict__ d2) {
    int b = blockIdx.x, o = threadIdx.x;               // 8 blocks x 256 threads
    float acc1 = 0.f;
    for (int c = 0; c < 512; ++c) { float sv = s1[b * 512 + c]; acc1 += wsq1[(size_t)o * 512 + c] * sv * sv; }
    d1[b * COUT + o] = rsqrtf(acc1 * (1.f / 4608.f) + 1e-8f);
    float acc2 = 0.f;
    for (int c = 0; c < 256; ++c) { float sv = s2[b * 256 + c]; acc2 += wsq2[(size_t)o * 256 + c] * sv * sv; }
    d2[b * COUT + o] = rsqrtf(acc2 * (1.f / 2304.f) + 1e-8f);
}

// ---------------------------------------------------------------------------
// Fused modulate (s1) + bilinear 2x upsample (half-pixel, edge clamp) -> f16
// ---------------------------------------------------------------------------
__global__ void upmod_kernel(const float* __restrict__ x, const float* __restrict__ s1,
                             _Float16* __restrict__ xup) {
    size_t idx = (size_t)blockIdx.x * 256 + threadIdx.x;   // [b][c][Y][X], 67,108,864 total
    int X = idx & 127, Y = (idx >> 7) & 127, c = (idx >> 14) & 511, b = (int)(idx >> 23);
    int ky = Y >> 1, kx = X >> 1;
    int y0, y1, x0, x1; float wy0, wy1, wx0, wx1;
    if (!(Y & 1)) { y0 = ky > 0 ? ky - 1 : 0; y1 = ky; wy0 = 0.25f; wy1 = 0.75f; }
    else          { y0 = ky; y1 = ky < 63 ? ky + 1 : 63; wy0 = 0.75f; wy1 = 0.25f; }
    if (!(X & 1)) { x0 = kx > 0 ? kx - 1 : 0; x1 = kx; wx0 = 0.25f; wx1 = 0.75f; }
    else          { x0 = kx; x1 = kx < 63 ? kx + 1 : 63; wx0 = 0.75f; wx1 = 0.25f; }
    const float* p = x + ((size_t)(b * 512 + c)) * 64 * 64;
    float v = wy0 * (wx0 * p[y0 * 64 + x0] + wx1 * p[y0 * 64 + x1])
            + wy1 * (wx0 * p[y1 * 64 + x0] + wx1 * p[y1 * 64 + x1]);
    xup[idx] = (_Float16)(v * s1[b * 512 + c]);
}

// ---------------------------------------------------------------------------
// Implicit-GEMM 3x3 modulated conv via WMMA f16->f32.
//   Block = one full output row: M=256 (cout) x N=128 (pixels), 512 thr = 16 waves
//   arranged 4(M) x 4(N); each wave owns 4x2 WMMA tiles (64 x 32).
// Per (dy, K-chunk of 32): stage B halo tile (130 cols) + A for all 3 dx taps
// into LDS once; all 16 waves consume via ds_load_b128 (A redundancy removed,
// weight L2 traffic ~4x lower than per-wave global fetch at N_block=64).
// Full-row tiles mean both halo columns are exactly the conv zero padding.
// Epilogue: *demod, +noise*nscale, leakyReLU(0.2); emit f32 (opt) + f16*snext.
// ---------------------------------------------------------------------------
template<int CIN>
__global__ __launch_bounds__(512)
void modconv3x3_wmma_kernel(const _Float16* __restrict__ in,     // [8][CIN][128][128]
                            const _Float16* __restrict__ wh,     // [9][256][CIN]
                            const float* __restrict__ demod,     // [8][256]
                            const float* __restrict__ snext,     // [8][256]
                            const float* __restrict__ noise,     // [8][128][128]
                            const float* __restrict__ nscale_p,  // [1]
                            float* __restrict__ out_f32,         // [8][256][128][128] or null
                            _Float16* __restrict__ out_f16)      // [8][256][128][128]
{
    constexpr int KPAD = 40;                        // 32 + 8 halves row pad (80 B rows)
    __shared__ _Float16 Bt[130 * KPAD];             // halo tile [col][k], 10,400 B
    __shared__ _Float16 As[3 * 256 * 32];           // A for 3 dx taps, 49,152 B
    __shared__ float dS[COUT], sS[COUT];

    const int tid = threadIdx.x;
    const int blk = blockIdx.x;                     // 1024 blocks
    const int b   = blk >> 7;
    const int y   = blk & 127;

    if (tid < COUT) { dS[tid] = demod[b * COUT + tid]; sS[tid] = snext[b * COUT + tid]; }
    const float ns = nscale_p[0];

    const int lane  = tid & 31;
    const int wid   = tid >> 5;                     // 0..15
    const int mBase = (wid & 3) << 6;               // 0,64,128,192
    const int nBase = (wid >> 2) << 5;              // 0,32,64,96
    const int lrow  = lane & 15;
    const int kb    = (lane >> 4) << 3;             // 0 or 8 (WMMA 16-bit lane K split)

    v8f acc[4][2];
#pragma unroll
    for (int i = 0; i < 4; ++i)
#pragma unroll
        for (int j = 0; j < 2; ++j) acc[i][j] = v8f{};

    const int cOff = tid >> 4;                      // 0..31: channel within K chunk
    const int seg  = (tid & 15) << 3;               // 0..120: pixel segment

    for (int dyi = 0; dyi < 3; ++dyi) {
        const int ys = y + dyi - 1;
        const bool rowok = (unsigned)ys < 128u;     // uniform across block
        const int  ysc   = rowok ? ys : 0;

        for (int kc = 0; kc < CIN; kc += 32) {
            __syncthreads();
            {   // ---- stage B halo tile: Bt[col][k], col 0..129 <-> xs = col-1
                const _Float16* row =
                    in + (((size_t)(b * CIN + kc + cOff)) * 128 + ysc) * 128;
                v8h val = v8h{};
                if (rowok) val = *(const v8h*)(row + seg);   // aligned b128
#pragma unroll
                for (int j = 0; j < 8; ++j)
                    Bt[(seg + 1 + j) * KPAD + cOff] = val[j];
                if (tid < 64)                                // zero-pad halo cols
                    Bt[((tid >> 5) ? 129 : 0) * KPAD + (tid & 31)] = (_Float16)0.f;
            }
            {   // ---- stage A for this dy's 3 taps: As[(tap*256+m)*32 + k2]
                const _Float16* wdy = wh + (size_t)(dyi * 3) * COUT * CIN;
#pragma unroll
                for (int r = 0; r < 6; ++r) {               // 3072 b128 transfers total
                    int idx   = r * 512 + tid;
                    int tapM  = idx >> 2;                   // 0..767 = tap*256+m
                    int j8    = (idx & 3) << 3;             // 0,8,16,24
                    v8h wv = *(const v8h*)(wdy + (size_t)tapM * CIN + kc + j8);
                    *(v8h*)&As[tapM * 32 + j8] = wv;        // ds_store_b128
                }
            }
            __syncthreads();

            // ---- 3 dx taps, A and B both from LDS
#pragma unroll
            for (int dxi = 0; dxi < 3; ++dxi) {
                v16h afr[4];
#pragma unroll
                for (int mi = 0; mi < 4; ++mi) {
                    const _Float16* ap =
                        &As[(dxi * 256 + mBase + mi * 16 + lrow) * 32 + kb];
                    v8h lo = *(const v8h*)ap;        // ds_load_b128
                    v8h hi = *(const v8h*)(ap + 16); // ds_load_b128
                    v16h f;
#pragma unroll
                    for (int e = 0; e < 8; ++e) { f[e] = lo[e]; f[e + 8] = hi[e]; }
                    afr[mi] = f;
                }
#pragma unroll
                for (int ni = 0; ni < 2; ++ni) {
                    // output col n needs input col (n + dxi) in halo coords
                    const _Float16* bp =
                        &Bt[(nBase + ni * 16 + lrow + dxi) * KPAD + kb];
                    v8h lo = *(const v8h*)bp;        // ds_load_b128
                    v8h hi = *(const v8h*)(bp + 16); // ds_load_b128
                    v16h bf;
#pragma unroll
                    for (int e = 0; e < 8; ++e) { bf[e] = lo[e]; bf[e + 8] = hi[e]; }
#pragma unroll
                    for (int mi = 0; mi < 4; ++mi) {
                        acc[mi][ni] = __builtin_amdgcn_wmma_f32_16x16x32_f16(
                            false, afr[mi], false, bf, (short)0, acc[mi][ni], false, false);
                    }
                }
            }
        }
    }
    __syncthreads();

    // Epilogue. D layout: VGPR r, lane L -> m = r + 8*(L>=16), n = L&15.
    const int hi8  = (lane >> 4) << 3;
    const int ncol = lane & 15;
#pragma unroll
    for (int mi = 0; mi < 4; ++mi) {
#pragma unroll
        for (int ni = 0; ni < 2; ++ni) {
            const int X  = nBase + ni * 16 + ncol;
            const float nv = ns * noise[((size_t)b * 128 + y) * 128 + X];
#pragma unroll
            for (int r = 0; r < 8; ++r) {
                const int m = mBase + mi * 16 + hi8 + r;
                float v = acc[mi][ni][r] * dS[m] + nv;
                v = v > 0.f ? v : 0.2f * v;                     // leaky ReLU
                const size_t oidx = (((size_t)b * COUT + m) * 128 + y) * 128 + X;
                if (out_f32) out_f32[oidx] = v;
                out_f16[oidx] = (_Float16)(v * sS[m]);          // fold next-conv style
            }
        }
    }
}

// ---------------------------------------------------------------------------
// to_rgb 1x1 conv (no demod, style pre-folded into h3) + bilinear skip add
// ---------------------------------------------------------------------------
__global__ void rgb_kernel(const _Float16* __restrict__ h3,   // [8][256][128][128] = h2*s3
                           const float* __restrict__ rgbw,    // [3][256]
                           const float* __restrict__ skip,    // [8][3][64][64]
                           float* __restrict__ out)           // [8][3][128][128]
{
    __shared__ float wS[3 * 256];
    int tid = threadIdx.x;
    for (int i = tid; i < 768; i += 256) wS[i] = rgbw[i] * 0.0625f;  // 1/sqrt(256)
    __syncthreads();
    size_t p = (size_t)blockIdx.x * 256 + tid;          // 131072 pixels
    int X = p & 127, Y = (int)((p >> 7) & 127), b = (int)(p >> 14);
    float a0 = 0.f, a1 = 0.f, a2 = 0.f;
    const _Float16* hb = h3 + ((size_t)b * 256) * 16384 + (Y << 7) + X;
    for (int c = 0; c < 256; ++c) {
        float hv = (float)hb[(size_t)c * 16384];        // coalesced across lanes
        a0 += hv * wS[c]; a1 += hv * wS[256 + c]; a2 += hv * wS[512 + c];
    }
    int ky = Y >> 1, kx = X >> 1;
    int y0, y1, x0, x1; float wy0, wy1, wx0, wx1;
    if (!(Y & 1)) { y0 = ky > 0 ? ky - 1 : 0; y1 = ky; wy0 = 0.25f; wy1 = 0.75f; }
    else          { y0 = ky; y1 = ky < 63 ? ky + 1 : 63; wy0 = 0.75f; wy1 = 0.25f; }
    if (!(X & 1)) { x0 = kx > 0 ? kx - 1 : 0; x1 = kx; wx0 = 0.25f; wx1 = 0.75f; }
    else          { x0 = kx; x1 = kx < 63 ? kx + 1 : 63; wx0 = 0.75f; wx1 = 0.25f; }
    float res[3] = {a0, a1, a2};
#pragma unroll
    for (int o = 0; o < 3; ++o) {
        const float* sp = skip + ((size_t)(b * 3 + o)) * 64 * 64;
        float sv = wy0 * (wx0 * sp[y0 * 64 + x0] + wx1 * sp[y0 * 64 + x1])
                 + wy1 * (wx0 * sp[y1 * 64 + x0] + wx1 * sp[y1 * 64 + x1]);
        out[(((size_t)b * 3 + o) << 14) + (Y << 7) + X] = res[o] + sv;
    }
}

// ---------------------------------------------------------------------------
extern "C" void kernel_launch(void* const* d_in, const int* in_sizes, int n_in,
                              void* d_out, int out_size, void* d_ws, size_t ws_size,
                              hipStream_t stream) {
    const float* x      = (const float*)d_in[0];
    const float* w1s    = (const float*)d_in[1];
    const float* w2s    = (const float*)d_in[2];
    const float* skip   = (const float*)d_in[3];
    const float* noise1 = (const float*)d_in[4];
    const float* noise2 = (const float*)d_in[5];
    const float* c1w    = (const float*)d_in[6];
    const float* m1w    = (const float*)d_in[7];
    const float* m1b    = (const float*)d_in[8];
    const float* c2w    = (const float*)d_in[9];
    const float* m2w    = (const float*)d_in[10];
    const float* m2b    = (const float*)d_in[11];
    const float* rgbw   = (const float*)d_in[12];
    const float* m3w    = (const float*)d_in[13];
    const float* m3b    = (const float*)d_in[14];
    const float* nw1    = (const float*)d_in[15];
    const float* nw2    = (const float*)d_in[16];

    float* out_h   = (float*)d_out;                              // [8,256,128,128]
    float* out_rgb = out_h + (size_t)8 * 256 * 128 * 128;        // [8,3,128,128]

    char* ws = (char*)d_ws;
    _Float16* xup  = (_Float16*)(ws + 0);                        // 134,217,728 B
    _Float16* h2s3 = (_Float16*)(ws + 0);                        // reuses xup (dead)
    _Float16* h1m  = (_Float16*)(ws + 134217728ull);             //  67,108,864 B
    _Float16* w1h  = (_Float16*)(ws + 201326592ull);             //   2,359,296 B
    _Float16* w2h  = (_Float16*)(ws + 203685888ull);             //   1,179,648 B
    float*    wsq1 = (float*)(ws + 204865536ull);                //     524,288 B
    float*    wsq2 = (float*)(ws + 205389824ull);                //     262,144 B
    float*    s1   = (float*)(ws + 205651968ull);
    float*    s2   = (float*)(ws + 205668352ull);
    float*    s3   = (float*)(ws + 205676544ull);
    float*    d1   = (float*)(ws + 205684736ull);
    float*    d2   = (float*)(ws + 205692928ull);

    const float sc1 = 1.f / sqrtf(512.f * 9.f);
    const float sc2 = 1.f / sqrtf(256.f * 9.f);

    prep_weights_kernel<512><<<512, 256, 0, stream>>>(c1w, w1h, wsq1, sc1);
    prep_weights_kernel<256><<<256, 256, 0, stream>>>(c2w, w2h, wsq2, sc2);
    styles_kernel<<<8, 256, 0, stream>>>(w1s, w2s, m1w, m1b, m2w, m2b, m3w, m3b, s1, s2, s3);
    demod_kernel<<<8, 256, 0, stream>>>(wsq1, wsq2, s1, s2, d1, d2);
    upmod_kernel<<<262144, 256, 0, stream>>>(x, s1, xup);
    modconv3x3_wmma_kernel<512><<<1024, 512, 0, stream>>>(
        xup, w1h, d1, s2, noise1, nw1, nullptr, h1m);
    modconv3x3_wmma_kernel<256><<<1024, 512, 0, stream>>>(
        h1m, w2h, d2, s3, noise2, nw2, out_h, h2s3);
    rgb_kernel<<<512, 256, 0, stream>>>(h2s3, rgbw, skip, out_rgb);
}
